// Block_60421599920798
// MI455X (gfx1250) — matrix operand order repaired and checked by reading, not compiled, not run on hardware
//
#include <hip/hip_runtime.h>
#include <math.h>

// ---------------------------------------------------------------------------
// RWKV-5 block for MI455X (gfx1250): bf16 WMMA GEMMs (async-to-LDS double
// buffered) + VALU WKV scan.
// ---------------------------------------------------------------------------

typedef __bf16 bf16;
typedef __bf16 v16bf __attribute__((ext_vector_type(16)));
typedef float  v8f   __attribute__((ext_vector_type(8)));

static constexpr int  Bb = 8, Tt = 2048, Cc = 2048, Hh = 32, HSz = 64, Ff = 7168;
static constexpr long BT   = (long)Bb * Tt;        // 16384 rows
static constexpr long BTC  = BT * Cc;              // 33,554,432 elems

// ---------------------------------------------------------------------------
// f32 -> bf16 conversion (weights, once per launch)
// ---------------------------------------------------------------------------
__global__ __launch_bounds__(256) void k_cvt_bf16(const float* __restrict__ src,
                                                  bf16* __restrict__ dst, long n) {
  long i = (long)blockIdx.x * 256 + threadIdx.x;
  if (i < n) dst[i] = (bf16)src[i];
}

// ---------------------------------------------------------------------------
// Fused LN0 + LN1: one block per (b,t) row of C=2048. Stores x0 and xn.
// ---------------------------------------------------------------------------
__global__ __launch_bounds__(256) void k_ln0_ln1(const float* __restrict__ X,
    const float* __restrict__ w0, const float* __restrict__ b0,
    const float* __restrict__ w1, const float* __restrict__ b1,
    float* __restrict__ X0, float* __restrict__ XN) {
  const int tid = threadIdx.x;
  const size_t row = blockIdx.x;
  const float* px = X + row * (size_t)Cc;
  float v[8];
#pragma unroll
  for (int i = 0; i < 8; ++i) v[i] = px[tid + i * 256];
  __shared__ float red[256];

  float s = 0.f;
#pragma unroll
  for (int i = 0; i < 8; ++i) s += v[i];
  red[tid] = s; __syncthreads();
  for (int o = 128; o > 0; o >>= 1) { if (tid < o) red[tid] += red[tid + o]; __syncthreads(); }
  const float m1 = red[0] * (1.f / Cc);
  __syncthreads();
  s = 0.f;
#pragma unroll
  for (int i = 0; i < 8; ++i) { float d = v[i] - m1; s += d * d; }
  red[tid] = s; __syncthreads();
  for (int o = 128; o > 0; o >>= 1) { if (tid < o) red[tid] += red[tid + o]; __syncthreads(); }
  const float rs1 = rsqrtf(red[0] * (1.f / Cc) + 1e-5f);
  __syncthreads();

  float y[8];
#pragma unroll
  for (int i = 0; i < 8; ++i) {
    int c = tid + i * 256;
    y[i] = (v[i] - m1) * rs1 * w0[c] + b0[c];
    X0[row * Cc + c] = y[i];
  }
  s = 0.f;
#pragma unroll
  for (int i = 0; i < 8; ++i) s += y[i];
  red[tid] = s; __syncthreads();
  for (int o = 128; o > 0; o >>= 1) { if (tid < o) red[tid] += red[tid + o]; __syncthreads(); }
  const float m2 = red[0] * (1.f / Cc);
  __syncthreads();
  s = 0.f;
#pragma unroll
  for (int i = 0; i < 8; ++i) { float d = y[i] - m2; s += d * d; }
  red[tid] = s; __syncthreads();
  for (int o = 128; o > 0; o >>= 1) { if (tid < o) red[tid] += red[tid + o]; __syncthreads(); }
  const float rs2 = rsqrtf(red[0] * (1.f / Cc) + 1e-5f);
#pragma unroll
  for (int i = 0; i < 8; ++i) {
    int c = tid + i * 256;
    XN[row * Cc + c] = (y[i] - m2) * rs2 * w1[c] + b1[c];
  }
}

// Single LN (for ln2)
__global__ __launch_bounds__(256) void k_ln(const float* __restrict__ X,
    const float* __restrict__ w, const float* __restrict__ b,
    float* __restrict__ Y) {
  const int tid = threadIdx.x;
  const size_t row = blockIdx.x;
  const float* px = X + row * (size_t)Cc;
  float v[8];
#pragma unroll
  for (int i = 0; i < 8; ++i) v[i] = px[tid + i * 256];
  __shared__ float red[256];
  float s = 0.f;
#pragma unroll
  for (int i = 0; i < 8; ++i) s += v[i];
  red[tid] = s; __syncthreads();
  for (int o = 128; o > 0; o >>= 1) { if (tid < o) red[tid] += red[tid + o]; __syncthreads(); }
  const float m = red[0] * (1.f / Cc);
  __syncthreads();
  s = 0.f;
#pragma unroll
  for (int i = 0; i < 8; ++i) { float d = v[i] - m; s += d * d; }
  red[tid] = s; __syncthreads();
  for (int o = 128; o > 0; o >>= 1) { if (tid < o) red[tid] += red[tid + o]; __syncthreads(); }
  const float rs = rsqrtf(red[0] * (1.f / Cc) + 1e-5f);
#pragma unroll
  for (int i = 0; i < 8; ++i) {
    int c = tid + i * 256;
    Y[row * Cc + c] = (v[i] - m) * rs * w[c] + b[c];
  }
}

// ---------------------------------------------------------------------------
// Attention token-shift mix: 4 bf16 outputs + attstate_n
// ---------------------------------------------------------------------------
__global__ __launch_bounds__(256) void k_mix_att(const float* __restrict__ XN,
    const float* __restrict__ st, const float* __restrict__ tmk,
    const float* __restrict__ tmv, const float* __restrict__ tmr,
    const float* __restrict__ tmg,
    bf16* __restrict__ xk, bf16* __restrict__ xv,
    bf16* __restrict__ xr, bf16* __restrict__ xg,
    float* __restrict__ st_n) {
  size_t i = (size_t)blockIdx.x * 256 + threadIdx.x;
  int c = (int)(i & (Cc - 1));
  size_t bt = i >> 11;
  int t = (int)(bt & (Tt - 1));
  int b = (int)(bt >> 11);
  float cur = XN[i];
  float prev = (t == 0) ? st[(size_t)b * Cc + c] : XN[i - Cc];
  float f;
  f = tmk[c]; xk[i] = (bf16)(cur * f + prev * (1.f - f));
  f = tmv[c]; xv[i] = (bf16)(cur * f + prev * (1.f - f));
  f = tmr[c]; xr[i] = (bf16)(cur * f + prev * (1.f - f));
  f = tmg[c]; xg[i] = (bf16)(cur * f + prev * (1.f - f));
  if (t == Tt - 1) st_n[(size_t)b * Cc + c] = cur;
}

// FFN token-shift mix: 2 bf16 outputs + ffnstate_n
__global__ __launch_bounds__(256) void k_mix_ffn(const float* __restrict__ XN,
    const float* __restrict__ st, const float* __restrict__ tmk,
    const float* __restrict__ tmr,
    bf16* __restrict__ xk, bf16* __restrict__ xr,
    float* __restrict__ st_n) {
  size_t i = (size_t)blockIdx.x * 256 + threadIdx.x;
  int c = (int)(i & (Cc - 1));
  size_t bt = i >> 11;
  int t = (int)(bt & (Tt - 1));
  int b = (int)(bt >> 11);
  float cur = XN[i];
  float prev = (t == 0) ? st[(size_t)b * Cc + c] : XN[i - Cc];
  float f;
  f = tmk[c]; xk[i] = (bf16)(cur * f + prev * (1.f - f));
  f = tmr[c]; xr[i] = (bf16)(cur * f + prev * (1.f - f));
  if (t == Tt - 1) st_n[(size_t)b * Cc + c] = cur;
}

// ---------------------------------------------------------------------------
// bf16 NT GEMM via v_wmma_f32_16x16x32_bf16.
//   C[m,n] = act( sum_k A[m,k] * Bw[n,k] )  [* Cmul] [+ Cadd]
// Block tile 128x128, K-step 32, 8 waves (2x4), each wave 64x32 (8 WMMA tiles).
// Staging: GLOBAL_LOAD_ASYNC_TO_LDS_B128 into double-buffered LDS tiles;
// wait on ASYNCcnt, one barrier per K-step, next tile's asyncs overlap WMMAs.
// ACT: 0=none 1=silu 2=relu^2 3=sigmoid
// ---------------------------------------------------------------------------
union Frag { v16bf v; uint4 u[2]; };

template<int ACT, bool BF16OUT, bool HASADD, bool HASMUL>
__global__ __launch_bounds__(256)
void k_gemm_nt(const bf16* __restrict__ A, const bf16* __restrict__ Bw,
               void* __restrict__ Cout, const float* __restrict__ Cadd,
               const float* __restrict__ Cmul, int M, int N, int K) {
  __shared__ __align__(16) bf16 lA[2][128 * 40];   // +8 bf16 row pad
  __shared__ __align__(16) bf16 lB[2][128 * 40];
  const int tid = threadIdx.x, lane = tid & 31, wid = tid >> 5;
  const int m0 = blockIdx.y * 128, n0 = blockIdx.x * 128;
  const int waveM = (wid >> 2) * 64, waveN = (wid & 3) * 32;

  v8f acc[4][2];
  const v8f vzero = {0.f, 0.f, 0.f, 0.f, 0.f, 0.f, 0.f, 0.f};
#pragma unroll
  for (int mt = 0; mt < 4; ++mt)
#pragma unroll
    for (int nt = 0; nt < 2; ++nt) acc[mt][nt] = vzero;

  // staging: 128 rows x 64B per tile; each thread moves 2x16B of A and of B
  const int r0 = tid >> 2, kc0 = (tid & 3) * 8;
  const int lr = lane & 15;
  const int aklo = (lane < 16) ? 0 : 8;    // A 16-bit 16x32 lane layout
  const int bklo = (lane < 16) ? 0 : 8;    // B 16-bit 32x16 lane layout

  // async DMA of one K-tile (4 x b128 per thread) directly into LDS buffer
  auto stage = [&](int buf, int kb) {
    unsigned long long ga0 = (unsigned long long)(A  + (size_t)(m0 + r0)      * K + kb + kc0);
    unsigned long long ga1 = (unsigned long long)(A  + (size_t)(m0 + r0 + 64) * K + kb + kc0);
    unsigned long long gb0 = (unsigned long long)(Bw + (size_t)(n0 + r0)      * K + kb + kc0);
    unsigned long long gb1 = (unsigned long long)(Bw + (size_t)(n0 + r0 + 64) * K + kb + kc0);
    unsigned la0 = (unsigned)(size_t)&lA[buf][r0 * 40 + kc0];
    unsigned la1 = (unsigned)(size_t)&lA[buf][(r0 + 64) * 40 + kc0];
    unsigned lb0 = (unsigned)(size_t)&lB[buf][r0 * 40 + kc0];
    unsigned lb1 = (unsigned)(size_t)&lB[buf][(r0 + 64) * 40 + kc0];
    asm volatile(
        "global_load_async_to_lds_b128 %0, %4, off\n\t"
        "global_load_async_to_lds_b128 %1, %5, off\n\t"
        "global_load_async_to_lds_b128 %2, %6, off\n\t"
        "global_load_async_to_lds_b128 %3, %7, off"
        :
        : "v"(la0), "v"(la1), "v"(lb0), "v"(lb1),
          "v"(ga0), "v"(ga1), "v"(gb0), "v"(gb1)
        : "memory");
  };

  stage(0, 0);
  for (int kb = 0; kb < K; kb += 32) {
    const int cur = (kb >> 5) & 1;
    asm volatile("s_wait_asynccnt 0x0" ::: "memory");  // tile `cur` landed in LDS
    __syncthreads();
    if (kb + 32 < K) stage(cur ^ 1, kb + 32);          // overlap next DMA w/ WMMAs

    Frag fa[4], fb[2];
#pragma unroll
    for (int mt = 0; mt < 4; ++mt) {
      const bf16* p = &lA[cur][(waveM + mt * 16 + lr) * 40 + aklo];
      fa[mt].u[0] = *(const uint4*)p;
      fa[mt].u[1] = *(const uint4*)(p + 16);
    }
#pragma unroll
    for (int nt = 0; nt < 2; ++nt) {
      const bf16* p = &lB[cur][(waveN + nt * 16 + lr) * 40 + bklo];
      fb[nt].u[0] = *(const uint4*)p;
      fb[nt].u[1] = *(const uint4*)(p + 16);
    }
#pragma unroll
    for (int mt = 0; mt < 4; ++mt)
#pragma unroll
      for (int nt = 0; nt < 2; ++nt)
        acc[mt][nt] = __builtin_amdgcn_wmma_f32_16x16x32_bf16(
            false, fa[mt].v, false, fb[nt].v, (short)0, acc[mt][nt], false, false);
  }

  // epilogue: C/D 16x16 f32 layout (VGPR i: M=i / M=i+8; lanes = N)
  const int col = lane & 15, rhalf = (lane >> 4) * 8;
#pragma unroll
  for (int mt = 0; mt < 4; ++mt) {
#pragma unroll
    for (int nt = 0; nt < 2; ++nt) {
      const int mb = m0 + waveM + mt * 16 + rhalf;
      const int nb = n0 + waveN + nt * 16 + col;
#pragma unroll
      for (int i = 0; i < 8; ++i) {
        size_t idx = (size_t)(mb + i) * N + nb;
        float val = acc[mt][nt][i];
        if (ACT == 1) val = val / (1.f + __expf(-val));           // silu
        if (ACT == 2) { val = fmaxf(val, 0.f); val *= val; }      // relu^2
        if (ACT == 3) val = 1.f / (1.f + __expf(-val));           // sigmoid
        if (HASMUL) val *= Cmul[idx];
        if (HASADD) val += Cadd[idx];
        if (BF16OUT) ((bf16*)Cout)[idx] = (bf16)val;
        else         ((float*)Cout)[idx] = val;
      }
    }
  }
}

// ---------------------------------------------------------------------------
// WKV scan: one block per (b,h). 256 threads: thread (g,j) holds S[i0..i0+15][j].
// Fuses the current-token term: out = r.S + (sum_i u_i r_i k_i) * v.
// ---------------------------------------------------------------------------
__global__ __launch_bounds__(256) void k_wkv_scan(const float* __restrict__ R,
    const float* __restrict__ Kk, const float* __restrict__ V,
    const float* __restrict__ wkv0, const float* __restrict__ tdec,
    const float* __restrict__ tfaa, float* __restrict__ OUT,
    float* __restrict__ wkv_n) {
  const int bh = blockIdx.x;
  const int b = bh >> 5, h = bh & 31;
  const int tid = threadIdx.x, j = tid & 63, g = tid >> 6, i0 = g * 16;
  __shared__ float sr[64], sk[64], sv[64];
  __shared__ float ly[4][64];
  __shared__ float lu[4];

  float S[16], wd[16], uu[16];
  const size_t hoff = (size_t)h * HSz;
#pragma unroll
  for (int ii = 0; ii < 16; ++ii) {
    wd[ii] = __expf(-__expf(tdec[hoff + i0 + ii]));
    uu[ii] = tfaa[hoff + i0 + ii];
    S[ii]  = wkv0[(((size_t)b * Hh + h) * HSz + (i0 + ii)) * HSz + j];
  }
  const size_t base = (size_t)b * Tt * Cc + hoff;
  for (int t = 0; t < Tt; ++t) {
    const size_t off = base + (size_t)t * Cc;
    __syncthreads();
    if (g == 0)      sr[j] = R[off + j];
    else if (g == 1) sk[j] = Kk[off + j];
    else if (g == 2) sv[j] = V[off + j];
    __syncthreads();
    float acc = 0.f, pu = 0.f;
#pragma unroll
    for (int ii = 0; ii < 16; ++ii) {
      float ri = sr[i0 + ii];
      acc += ri * S[ii];
      pu  += uu[ii] * ri * sk[i0 + ii];
    }
    ly[g][j] = acc;
    if (j == 0) lu[g] = pu;
    __syncthreads();
    if (g == 0) {
      float y   = ly[0][j] + ly[1][j] + ly[2][j] + ly[3][j];
      float urk = lu[0] + lu[1] + lu[2] + lu[3];
      OUT[off + j] = y + urk * sv[j];
    }
    const float vj = sv[j];
#pragma unroll
    for (int ii = 0; ii < 16; ++ii)
      S[ii] = S[ii] * wd[ii] + sk[i0 + ii] * vj;
  }
#pragma unroll
  for (int ii = 0; ii < 16; ++ii)
    wkv_n[(((size_t)b * Hh + h) * HSz + (i0 + ii)) * HSz + j] = S[ii];
}

// ---------------------------------------------------------------------------
// Per-head groupnorm (over HS=64) of out/8, scale by lnx, multiply by g -> bf16
// Block = 4 heads x 64 threads.
// ---------------------------------------------------------------------------
__global__ __launch_bounds__(256) void k_gn_g(const float* __restrict__ OUT,
    const float* __restrict__ G, const float* __restrict__ lw,
    const float* __restrict__ lb, bf16* __restrict__ OG) {
  const int tid = threadIdx.x, grp = tid >> 6, j = tid & 63;
  const long bth = (long)blockIdx.x * 4 + grp;
  const int h = (int)(bth & (Hh - 1));
  const size_t bt = (size_t)(bth >> 5);
  const size_t idx = bt * Cc + (size_t)h * HSz + j;
  float val = OUT[idx] * 0.125f;
  __shared__ float red[256];
  red[tid] = val; __syncthreads();
  for (int o = 32; o > 0; o >>= 1) { if (j < o) red[tid] += red[tid + o]; __syncthreads(); }
  const float m = red[grp * 64] * (1.f / 64.f);
  __syncthreads();
  const float d = val - m;
  red[tid] = d * d; __syncthreads();
  for (int o = 32; o > 0; o >>= 1) { if (j < o) red[tid] += red[tid + o]; __syncthreads(); }
  const float var = red[grp * 64] * (1.f / 64.f);
  const float o = d * rsqrtf(var + 1e-5f) * lw[h * HSz + j] + lb[h * HSz + j];
  OG[idx] = (bf16)(o * G[idx]);
}

// ---------------------------------------------------------------------------
// Host side
// ---------------------------------------------------------------------------
extern "C" void kernel_launch(void* const* d_in, const int* in_sizes, int n_in,
                              void* d_out, int out_size, void* d_ws, size_t ws_size,
                              hipStream_t stream) {
  (void)in_sizes; (void)n_in; (void)out_size; (void)ws_size;
  const float* x        = (const float*)d_in[0];
  const float* attstate = (const float*)d_in[1];
  const float* ffnstate = (const float*)d_in[2];
  const float* wkvstate = (const float*)d_in[3];
  const float* ln0_w = (const float*)d_in[4];
  const float* ln0_b = (const float*)d_in[5];
  const float* ln1_w = (const float*)d_in[6];
  const float* ln1_b = (const float*)d_in[7];
  const float* ln2_w = (const float*)d_in[8];
  const float* ln2_b = (const float*)d_in[9];
  const float* tm_k  = (const float*)d_in[10];
  const float* tm_v  = (const float*)d_in[11];
  const float* tm_r  = (const float*)d_in[12];
  const float* tm_g  = (const float*)d_in[13];
  const float* tdec  = (const float*)d_in[14];
  const float* tfaa  = (const float*)d_in[15];
  const float* Wg    = (const float*)d_in[16];
  const float* Wr    = (const float*)d_in[17];
  const float* Wk    = (const float*)d_in[18];
  const float* Wv    = (const float*)d_in[19];
  const float* Wo    = (const float*)d_in[20];
  const float* lnx_w = (const float*)d_in[21];
  const float* lnx_b = (const float*)d_in[22];
  const float* f_tmk = (const float*)d_in[23];
  const float* f_tmr = (const float*)d_in[24];
  const float* f_Wk  = (const float*)d_in[25];
  const float* f_Wr  = (const float*)d_in[26];
  const float* f_Wv  = (const float*)d_in[27];

  char* ws = (char*)d_ws;
  float* outx = (float*)d_out;                    // [B,T,C]
  float* out_ffn = outx + BTC;                    // [B,1,C]
  float* out_att = out_ffn + (size_t)Bb * Cc;     // [B,1,C]
  float* out_wkv = out_att + (size_t)Bb * Cc;     // [B,H,HS,HS]

  const size_t F32BTC = (size_t)BTC * 4, BF16BTC = (size_t)BTC * 2;
  const size_t CC_BF = (size_t)Cc * Cc * 2, FC_BF = (size_t)Ff * Cc * 2;
  const size_t O_X0 = 0;
  const size_t O_XN = O_X0 + F32BTC;
  const size_t O_XR = O_XN + F32BTC;
  const size_t O_XK = O_XR + BF16BTC;
  const size_t O_XV = O_XK + BF16BTC;
  const size_t O_XG = O_XV + BF16BTC;
  const size_t O_R  = O_XG + BF16BTC;
  const size_t O_K  = O_R + F32BTC;
  const size_t O_V  = O_K + F32BTC;
  const size_t O_G  = O_V + F32BTC;
  const size_t O_OUT = O_G + F32BTC;
  const size_t O_W   = O_OUT + F32BTC;
  const size_t O_Wrb = O_W, O_Wkb = O_Wrb + CC_BF, O_Wvb = O_Wkb + CC_BF;
  const size_t O_Wgb = O_Wvb + CC_BF, O_Wob = O_Wgb + CC_BF;
  const size_t O_fWkb = O_Wob + CC_BF, O_fWrb = O_fWkb + FC_BF, O_fWvb = O_fWrb + CC_BF;
  // reuse after attention phase:
  const size_t O_OG = O_XG;         // bf16 [BT,C]
  const size_t O_XK2 = O_XR, O_XR2 = O_XK;
  const size_t O_KK = O_R;          // bf16 [BT,F] (235MB fits in R+K span)
  const size_t O_R2 = O_V;          // f32 [BT,C]

  const long nCC = (long)Cc * Cc, nFC = (long)Ff * Cc;
  dim3 blk(256);
  // --- weight conversion ---
  k_cvt_bf16<<<(nCC + 255) / 256, blk, 0, stream>>>(Wr, (bf16*)(ws + O_Wrb), nCC);
  k_cvt_bf16<<<(nCC + 255) / 256, blk, 0, stream>>>(Wk, (bf16*)(ws + O_Wkb), nCC);
  k_cvt_bf16<<<(nCC + 255) / 256, blk, 0, stream>>>(Wv, (bf16*)(ws + O_Wvb), nCC);
  k_cvt_bf16<<<(nCC + 255) / 256, blk, 0, stream>>>(Wg, (bf16*)(ws + O_Wgb), nCC);
  k_cvt_bf16<<<(nCC + 255) / 256, blk, 0, stream>>>(Wo, (bf16*)(ws + O_Wob), nCC);
  k_cvt_bf16<<<(nFC + 255) / 256, blk, 0, stream>>>(f_Wk, (bf16*)(ws + O_fWkb), nFC);
  k_cvt_bf16<<<(nCC + 255) / 256, blk, 0, stream>>>(f_Wr, (bf16*)(ws + O_fWrb), nCC);
  k_cvt_bf16<<<(nFC + 255) / 256, blk, 0, stream>>>(f_Wv, (bf16*)(ws + O_fWvb), nFC);

  // --- attention phase ---
  k_ln0_ln1<<<(int)BT, blk, 0, stream>>>(x, ln0_w, ln0_b, ln1_w, ln1_b,
                                         (float*)(ws + O_X0), (float*)(ws + O_XN));
  k_mix_att<<<(int)(BTC / 256), blk, 0, stream>>>((const float*)(ws + O_XN), attstate,
      tm_k, tm_v, tm_r, tm_g, (bf16*)(ws + O_XK), (bf16*)(ws + O_XV),
      (bf16*)(ws + O_XR), (bf16*)(ws + O_XG), out_att);

  dim3 gCC(Cc / 128, (int)(BT / 128));    // (16,128)
  k_gemm_nt<0, false, false, false><<<gCC, blk, 0, stream>>>(
      (const bf16*)(ws + O_XR), (const bf16*)(ws + O_Wrb), ws + O_R, nullptr, nullptr,
      (int)BT, Cc, Cc);
  k_gemm_nt<0, false, false, false><<<gCC, blk, 0, stream>>>(
      (const bf16*)(ws + O_XK), (const bf16*)(ws + O_Wkb), ws + O_K, nullptr, nullptr,
      (int)BT, Cc, Cc);
  k_gemm_nt<0, false, false, false><<<gCC, blk, 0, stream>>>(
      (const bf16*)(ws + O_XV), (const bf16*)(ws + O_Wvb), ws + O_V, nullptr, nullptr,
      (int)BT, Cc, Cc);
  k_gemm_nt<1, false, false, false><<<gCC, blk, 0, stream>>>(   // silu
      (const bf16*)(ws + O_XG), (const bf16*)(ws + O_Wgb), ws + O_G, nullptr, nullptr,
      (int)BT, Cc, Cc);

  k_wkv_scan<<<Bb * Hh, blk, 0, stream>>>((const float*)(ws + O_R),
      (const float*)(ws + O_K), (const float*)(ws + O_V), wkvstate, tdec, tfaa,
      (float*)(ws + O_OUT), out_wkv);

  k_gn_g<<<(int)(BT * Hh / 4), blk, 0, stream>>>((const float*)(ws + O_OUT),
      (const float*)(ws + O_G), lnx_w, lnx_b, (bf16*)(ws + O_OG));

  // x1 = x0 + (og @ Wo^T)  -> d_out.x
  k_gemm_nt<0, false, true, false><<<gCC, blk, 0, stream>>>(
      (const bf16*)(ws + O_OG), (const bf16*)(ws + O_Wob), outx,
      (const float*)(ws + O_X0), nullptr, (int)BT, Cc, Cc);

  // --- FFN phase ---
  k_ln<<<(int)BT, blk, 0, stream>>>(outx, ln2_w, ln2_b, (float*)(ws + O_XN));
  k_mix_ffn<<<(int)(BTC / 256), blk, 0, stream>>>((const float*)(ws + O_XN), ffnstate,
      f_tmk, f_tmr, (bf16*)(ws + O_XK2), (bf16*)(ws + O_XR2), out_ffn);

  dim3 gFC(Ff / 128, (int)(BT / 128));    // (56,128)
  k_gemm_nt<2, true, false, false><<<gFC, blk, 0, stream>>>(  // relu^2 -> bf16 kk
      (const bf16*)(ws + O_XK2), (const bf16*)(ws + O_fWkb), ws + O_KK, nullptr, nullptr,
      (int)BT, Ff, Cc);
  k_gemm_nt<3, false, false, false><<<gCC, blk, 0, stream>>>( // sigmoid r2
      (const bf16*)(ws + O_XR2), (const bf16*)(ws + O_fWrb), ws + O_R2, nullptr, nullptr,
      (int)BT, Cc, Cc);
  // x = x1 + r2 * (kk @ f_Wv^T)  -> d_out.x (in place add)
  k_gemm_nt<0, false, true, true><<<gCC, blk, 0, stream>>>(
      (const bf16*)(ws + O_KK), (const bf16*)(ws + O_fWvb), outx,
      outx, (const float*)(ws + O_R2), (int)BT, Cc, Ff);
}
